// MixtureOfExpertsLayer_8538394984715
// MI455X (gfx1250) — matrix operand order repaired and checked by reading, not compiled, not run on hardware
//
#include <hip/hip_runtime.h>
#include <hip/hip_bf16.h>
#include <math.h>

// Problem constants (match reference)
#define BB 4
#define SS 2048
#define HH 1024
#define EE 8
#define KK 2
#define FF 4096
#define TT (BB * SS)          // 8192 tokens
#define NROWS (TT * KK)       // 16384 routed rows

// GEMM tiling
#define KCH  64               // K-chunk staged per TDM transfer
#define COLS 512              // N columns per block (8 waves x 64)

typedef __attribute__((ext_vector_type(16))) __bf16 v16bf;
typedef __attribute__((ext_vector_type(8)))  __bf16 v8bf;
typedef __attribute__((ext_vector_type(4)))  __bf16 v4bf;
typedef __attribute__((ext_vector_type(8)))  float  v8f;
typedef __attribute__((ext_vector_type(4)))  unsigned int v4u;
typedef __attribute__((ext_vector_type(8)))  int v8i;
typedef __attribute__((ext_vector_type(4)))  int v4i;

// Toolchain hedge: therock headers ship the 6-arg TDM builtin, ROCm 7.2 the 5-arg.
#if defined(__has_include)
#  if __has_include(<hip/amd_detail/amd_gfx1250_TDM.h>)
#    define MOE_TDM_6ARG 1
#  else
#    define MOE_TDM_6ARG 0
#  endif
#else
#  define MOE_TDM_6ARG 0
#endif

// ---------------------------------------------------------------------------
// Tensor Data Mover: 2D tile (optionally 16-row gather) global -> LDS.
// D# bitfields per CDNA5 ISA ch.8: group0 {count|gather, lds_addr, global_addr,
// type=2}; group1 {data_size=2B, tensor dims, tile dims, dim0 stride};
// groups 2/3 carry 16-bit row indices in gather mode.
// ---------------------------------------------------------------------------
__device__ __forceinline__ void tdm_load_2d(unsigned ldsAddr, const void* gptr,
                                            unsigned tensorD0, unsigned tensorD1,
                                            unsigned tileD0, unsigned tileD1,
                                            unsigned long long strideD0,
                                            int gather, const unsigned* gidx) {
    unsigned long long ga = (unsigned long long)(uintptr_t)gptr;
    v4u g0;
    g0[0] = 1u | (gather ? (1u << 31) : 0u);          // count=1 [, gather_mode, 16b idx]
    g0[1] = ldsAddr;                                  // LDS byte address
    g0[2] = (unsigned)ga;                             // global addr [31:0]
    g0[3] = (unsigned)((ga >> 32) & 0x01FFFFFFu) | (2u << 30);  // addr[56:32] | type=2
    v8i g1;
    g1[0] = (int)(1u << 16);                                          // data_size = 2B
    g1[1] = (int)((tensorD0 & 0xFFFFu) << 16);                        // dim0[15:0]
    g1[2] = (int)(((tensorD0 >> 16) & 0xFFFFu) | ((tensorD1 & 0xFFFFu) << 16));
    g1[3] = (int)(((tensorD1 >> 16) & 0xFFFFu) | ((tileD0 & 0xFFFFu) << 16));
    g1[4] = (int)(tileD1 & 0xFFFFu);                                  // tile_dim1 (|tile_dim2=0)
    g1[5] = (int)(unsigned)(strideD0 & 0xFFFFFFFFu);                  // dim0 stride lo
    g1[6] = (int)(unsigned)((strideD0 >> 32) & 0xFFFFu);              // stride hi | dim1 stride=0
    g1[7] = 0;
    v4i g2 = {}, g3 = {};
    if (gather) {
        g2[0] = (int)gidx[0]; g2[1] = (int)gidx[1]; g2[2] = (int)gidx[2]; g2[3] = (int)gidx[3];
        g3[0] = (int)gidx[4]; g3[1] = (int)gidx[5]; g3[2] = (int)gidx[6]; g3[3] = (int)gidx[7];
    }
#if MOE_TDM_6ARG
    v8i g4 = {};
    __builtin_amdgcn_tensor_load_to_lds(g0, g1, g2, g3, g4, 0);
#else
    __builtin_amdgcn_tensor_load_to_lds(g0, g1, g2, g3, 0);
#endif
}

// ---------------------------------------------------------------------------
// init: zero counters / usage
// ---------------------------------------------------------------------------
__global__ void moe_init(int* cnt, float* usage) {
    if (threadIdx.x < EE) { cnt[threadIdx.x] = 0; usage[threadIdx.x] = 0.f; }
}

// ---------------------------------------------------------------------------
// gate: one wave32 per token
// ---------------------------------------------------------------------------
__global__ __launch_bounds__(256) void moe_gate(const float* __restrict__ x,
                                                const float* __restrict__ gw,
                                                int* __restrict__ cnt,
                                                float* __restrict__ usage,
                                                int2* __restrict__ tk_idx,
                                                float2* __restrict__ tk_w) {
    __shared__ float su[EE];
    const int lane = threadIdx.x & 31;
    const int wave = threadIdx.x >> 5;
    const int t = blockIdx.x * 8 + wave;

    if (threadIdx.x < EE) su[threadIdx.x] = 0.f;
    __syncthreads();

    float acc[EE];
#pragma unroll
    for (int e = 0; e < EE; ++e) acc[e] = 0.f;

    const float* xr = x + (size_t)t * HH;
    for (int h = lane; h < HH; h += 32) {
        float xv = xr[h];
        const float* g = gw + (size_t)h * EE;
#pragma unroll
        for (int e = 0; e < EE; ++e) acc[e] += xv * g[e];
    }
#pragma unroll
    for (int e = 0; e < EE; ++e) {
#pragma unroll
        for (int s = 16; s > 0; s >>= 1) acc[e] += __shfl_xor(acc[e], s, 32);
    }

    float mx = acc[0];
#pragma unroll
    for (int e = 1; e < EE; ++e) mx = fmaxf(mx, acc[e]);
    float p[EE], sum = 0.f;
#pragma unroll
    for (int e = 0; e < EE; ++e) { p[e] = __expf(acc[e] - mx); sum += p[e]; }
    float inv = 1.f / sum;
#pragma unroll
    for (int e = 0; e < EE; ++e) p[e] *= inv;

    int i0 = 0; float v0 = p[0];
#pragma unroll
    for (int e = 1; e < EE; ++e) if (p[e] > v0) { v0 = p[e]; i0 = e; }
    int i1 = (i0 == 0) ? 1 : 0; float v1 = p[i1];
#pragma unroll
    for (int e = 0; e < EE; ++e)
        if (e != i0 && p[e] > v1) { v1 = p[e]; i1 = e; }

    float b = __expf(v1 - v0);
    float w0 = 1.f / (1.f + b);
    float w1 = 1.f - w0;

    if (lane == 0) {
        tk_idx[t] = make_int2(i0, i1);
        tk_w[t]   = make_float2(w0, w1);
        atomicAdd(&cnt[i0], 1);
        atomicAdd(&cnt[i1], 1);
#pragma unroll
        for (int e = 0; e < EE; ++e) atomicAdd(&su[e], p[e]);
    }
    __syncthreads();
    if (threadIdx.x < EE) atomicAdd(&usage[threadIdx.x], su[threadIdx.x]);
}

// ---------------------------------------------------------------------------
// prefix + scatter (contiguous per-expert row ranges, deterministic combine)
// ---------------------------------------------------------------------------
__global__ void moe_prefix(const int* __restrict__ cnt, int* __restrict__ off,
                           int* __restrict__ cur) {
    if (threadIdx.x == 0) {
        int o = 0;
        for (int e = 0; e < EE; ++e) { off[e] = o; cur[e] = o; o += cnt[e]; }
    }
}

__global__ __launch_bounds__(256) void moe_scatter(const int2* __restrict__ tk_idx,
                                                   int* __restrict__ cur,
                                                   int* __restrict__ list,
                                                   int2* __restrict__ slot_row) {
    int t = blockIdx.x * 256 + threadIdx.x;
    if (t >= TT) return;
    int2 idx = tk_idx[t];
    int g0 = atomicAdd(&cur[idx.x], 1);
    int g1 = atomicAdd(&cur[idx.y], 1);
    list[g0] = t;
    list[g1] = t;
    slot_row[t] = make_int2(g0, g1);
}

// ---------------------------------------------------------------------------
// conversions
// ---------------------------------------------------------------------------
__global__ __launch_bounds__(256) void cvt_x(const float4* __restrict__ in,
                                             v4bf* __restrict__ out) {
    int i = blockIdx.x * 256 + threadIdx.x;
    float4 v = in[i];
    v4bf o;
    o[0] = (__bf16)v.x; o[1] = (__bf16)v.y; o[2] = (__bf16)v.z; o[3] = (__bf16)v.w;
    out[i] = o;
}

__global__ __launch_bounds__(256) void tcvt(const float* __restrict__ in,
                                            __bf16* __restrict__ out,
                                            int R, int C) {
    __shared__ float tile[32][33];
    const int bz = blockIdx.z;
    const int c0 = blockIdx.x * 32, r0 = blockIdx.y * 32;
    const float* src = in + (size_t)bz * R * C;
    __bf16* dst = out + (size_t)bz * R * C;
#pragma unroll
    for (int i = 0; i < 32; i += 8) {
        int r = r0 + threadIdx.y + i;
        tile[threadIdx.y + i][threadIdx.x] = src[(size_t)r * C + c0 + threadIdx.x];
    }
    __syncthreads();
#pragma unroll
    for (int i = 0; i < 32; i += 8) {
        int c = c0 + threadIdx.y + i;
        dst[(size_t)c * R + r0 + threadIdx.x] = (__bf16)tile[threadIdx.x][threadIdx.y + i];
    }
}

// ---------------------------------------------------------------------------
// GEMM1: h[row,F] = relu(x[list[row]] @ w1^T[e] + b1[e])
// TDM-gather A tile (16 routed rows), TDM B slab, double-buffered LDS,
// 8 waves x (16x64) register tiles -> block tile 16 x 512.
// ---------------------------------------------------------------------------
__global__ __launch_bounds__(256) void moe_gemm1(const __bf16* __restrict__ xb,
                                                 const __bf16* __restrict__ w1t,
                                                 const float* __restrict__ b1,
                                                 const int* __restrict__ list,
                                                 const int* __restrict__ cnt,
                                                 const int* __restrict__ off,
                                                 __bf16* __restrict__ hbuf) {
    __shared__ __bf16 aLds[2][16 * KCH];       // 2 x 2 KB
    __shared__ __bf16 bLds[2][COLS * KCH];     // 2 x 64 KB
    __shared__ float  tLds[8][16][17];

    const int e = blockIdx.z;
    const int c = cnt[e];
    const int rbase = blockIdx.y * 16;
    if (rbase >= c) return;
    const int o = off[e];

    const int lane = threadIdx.x & 31;
    const int wave = threadIdx.x >> 5;
    const int sel  = lane >> 4;
    const int q    = lane & 15;
    const int colblk = blockIdx.x * COLS;

    // routed row indices for this block (clamped tail), packed to 16-bit pairs
    int rr = rbase + q; if (rr >= c) rr = c - 1;
    int tok = list[o + rr];
    unsigned idxw[8];
#pragma unroll
    for (int i = 0; i < 8; ++i) {
        unsigned lo = (unsigned)__builtin_amdgcn_readlane(tok, 2 * i) & 0xFFFFu;
        unsigned hi = (unsigned)__builtin_amdgcn_readlane(tok, 2 * i + 1) & 0xFFFFu;
        idxw[i] = lo | (hi << 16);
    }

    const __bf16* bBase = w1t + ((size_t)e * FF + colblk) * HH;
    const int NCH = HH / KCH;   // 16

    if (wave == 0) {
        tdm_load_2d((unsigned)(uintptr_t)&aLds[0][0], xb, HH, TT, KCH, 16,
                    (unsigned long long)HH, 1, idxw);
        tdm_load_2d((unsigned)(uintptr_t)&bLds[0][0], bBase, HH, (unsigned)EE * FF,
                    KCH, COLS, (unsigned long long)HH, 0, 0);
    }

    v8f acc[4] = {};
    for (int ch = 0; ch < NCH; ++ch) {
        const int cur = ch & 1;
        if (wave == 0) {
            if (ch + 1 < NCH) {
                tdm_load_2d((unsigned)(uintptr_t)&aLds[cur ^ 1][0], xb + (ch + 1) * KCH,
                            HH, TT, KCH, 16, (unsigned long long)HH, 1, idxw);
                tdm_load_2d((unsigned)(uintptr_t)&bLds[cur ^ 1][0], bBase + (ch + 1) * KCH,
                            HH, (unsigned)EE * FF, KCH, COLS, (unsigned long long)HH, 0, 0);
                __builtin_amdgcn_s_wait_tensorcnt(2);
            } else {
                __builtin_amdgcn_s_wait_tensorcnt(0);
            }
        }
        __syncthreads();                               // chunk `ch` visible in LDS

        const __bf16* aB = &aLds[cur][0];
        const __bf16* bB = &bLds[cur][0];
#pragma unroll
        for (int kk = 0; kk < KCH; kk += 32) {
            v8bf a0 = *(const v8bf*)(aB + q * KCH + kk + sel * 8);
            v8bf a1 = *(const v8bf*)(aB + q * KCH + kk + 16 + sel * 8);
            v16bf av = __builtin_shufflevector(a0, a1, 0,1,2,3,4,5,6,7,8,9,10,11,12,13,14,15);
#pragma unroll
            for (int nt = 0; nt < 4; ++nt) {
                const int colL = wave * 64 + nt * 16 + q;
                v8bf b0 = *(const v8bf*)(bB + colL * KCH + kk + sel * 8);
                v8bf b1v = *(const v8bf*)(bB + colL * KCH + kk + 16 + sel * 8);
                v16bf bv = __builtin_shufflevector(b0, b1v, 0,1,2,3,4,5,6,7,8,9,10,11,12,13,14,15);
                acc[nt] = __builtin_amdgcn_wmma_f32_16x16x32_bf16(false, av, false, bv,
                                                                  (short)0, acc[nt], false, false);
            }
        }
        __syncthreads();                               // done reading before overwrite
    }

    // epilogue: bias + relu, per-wave LDS transpose -> contiguous 16B bf16 stores
#pragma unroll
    for (int nt = 0; nt < 4; ++nt) {
        const int colb = colblk + wave * 64 + nt * 16;
        float bias = b1[(size_t)e * FF + colb + q];
#pragma unroll
        for (int v = 0; v < 8; ++v) {
            float r = acc[nt][v] + bias;
            tLds[wave][v + 8 * sel][q] = r > 0.f ? r : 0.f;
        }
        __syncthreads();
        if (rbase + q < c) {
            v8bf ov;
#pragma unroll
            for (int j = 0; j < 8; ++j) ov[j] = (__bf16)tLds[wave][q][sel * 8 + j];
            *(v8bf*)(hbuf + (size_t)(o + rbase + q) * FF + colb + sel * 8) = ov;
        }
        __syncthreads();
    }
}

// ---------------------------------------------------------------------------
// GEMM2: eout[row,H] = h[row] @ w2^T[e] + b2[e]   (same structure, plain A tile)
// ---------------------------------------------------------------------------
__global__ __launch_bounds__(256) void moe_gemm2(const __bf16* __restrict__ hbuf,
                                                 const __bf16* __restrict__ w2t,
                                                 const float* __restrict__ b2,
                                                 const int* __restrict__ cnt,
                                                 const int* __restrict__ off,
                                                 float* __restrict__ eout) {
    __shared__ __bf16 aLds[2][16 * KCH];
    __shared__ __bf16 bLds[2][COLS * KCH];
    __shared__ float  tLds[8][16][17];

    const int e = blockIdx.z;
    const int c = cnt[e];
    const int rbase = blockIdx.y * 16;
    if (rbase >= c) return;
    const int o = off[e];

    const int lane = threadIdx.x & 31;
    const int wave = threadIdx.x >> 5;
    const int sel  = lane >> 4;
    const int q    = lane & 15;
    const int colblk = blockIdx.x * COLS;

    const __bf16* aBase = hbuf + (size_t)(o + rbase) * FF;
    const __bf16* bBase = w2t + ((size_t)e * HH + colblk) * FF;
    const int NCH = FF / KCH;   // 64

    if (wave == 0) {
        tdm_load_2d((unsigned)(uintptr_t)&aLds[0][0], aBase, FF, NROWS, KCH, 16,
                    (unsigned long long)FF, 0, 0);
        tdm_load_2d((unsigned)(uintptr_t)&bLds[0][0], bBase, FF, (unsigned)EE * HH,
                    KCH, COLS, (unsigned long long)FF, 0, 0);
    }

    v8f acc[4] = {};
    for (int ch = 0; ch < NCH; ++ch) {
        const int cur = ch & 1;
        if (wave == 0) {
            if (ch + 1 < NCH) {
                tdm_load_2d((unsigned)(uintptr_t)&aLds[cur ^ 1][0], aBase + (ch + 1) * KCH,
                            FF, NROWS, KCH, 16, (unsigned long long)FF, 0, 0);
                tdm_load_2d((unsigned)(uintptr_t)&bLds[cur ^ 1][0], bBase + (ch + 1) * KCH,
                            FF, (unsigned)EE * HH, KCH, COLS, (unsigned long long)FF, 0, 0);
                __builtin_amdgcn_s_wait_tensorcnt(2);
            } else {
                __builtin_amdgcn_s_wait_tensorcnt(0);
            }
        }
        __syncthreads();

        const __bf16* aB = &aLds[cur][0];
        const __bf16* bB = &bLds[cur][0];
#pragma unroll
        for (int kk = 0; kk < KCH; kk += 32) {
            v8bf a0 = *(const v8bf*)(aB + q * KCH + kk + sel * 8);
            v8bf a1 = *(const v8bf*)(aB + q * KCH + kk + 16 + sel * 8);
            v16bf av = __builtin_shufflevector(a0, a1, 0,1,2,3,4,5,6,7,8,9,10,11,12,13,14,15);
#pragma unroll
            for (int nt = 0; nt < 4; ++nt) {
                const int colL = wave * 64 + nt * 16 + q;
                v8bf b0 = *(const v8bf*)(bB + colL * KCH + kk + sel * 8);
                v8bf b1v = *(const v8bf*)(bB + colL * KCH + kk + 16 + sel * 8);
                v16bf bv = __builtin_shufflevector(b0, b1v, 0,1,2,3,4,5,6,7,8,9,10,11,12,13,14,15);
                acc[nt] = __builtin_amdgcn_wmma_f32_16x16x32_bf16(false, av, false, bv,
                                                                  (short)0, acc[nt], false, false);
            }
        }
        __syncthreads();
    }

#pragma unroll
    for (int nt = 0; nt < 4; ++nt) {
        const int colb = colblk + wave * 64 + nt * 16;
        float bias = b2[(size_t)e * HH + colb + q];
#pragma unroll
        for (int v = 0; v < 8; ++v) tLds[wave][v + 8 * sel][q] = acc[nt][v] + bias;
        __syncthreads();
        if (rbase + q < c) {
            float* dst = eout + (size_t)(o + rbase + q) * HH + colb + sel * 8;
            float4 lo, hi;
            lo.x = tLds[wave][q][sel * 8 + 0]; lo.y = tLds[wave][q][sel * 8 + 1];
            lo.z = tLds[wave][q][sel * 8 + 2]; lo.w = tLds[wave][q][sel * 8 + 3];
            hi.x = tLds[wave][q][sel * 8 + 4]; hi.y = tLds[wave][q][sel * 8 + 5];
            hi.z = tLds[wave][q][sel * 8 + 6]; hi.w = tLds[wave][q][sel * 8 + 7];
            ((float4*)dst)[0] = lo;
            ((float4*)dst)[1] = hi;
        }
        __syncthreads();
    }
}

// ---------------------------------------------------------------------------
// combine + loss
// ---------------------------------------------------------------------------
__global__ __launch_bounds__(256) void moe_combine(const float* __restrict__ eout,
                                                   const int2* __restrict__ slot_row,
                                                   const float2* __restrict__ tk_w,
                                                   float4* __restrict__ out) {
    int idx = blockIdx.x * 256 + threadIdx.x;
    int t = idx >> 8;
    int s = idx & 255;
    int2 rr = slot_row[t];
    float2 ww = tk_w[t];
    float4 a = ((const float4*)(eout + (size_t)rr.x * HH))[s];
    float4 b = ((const float4*)(eout + (size_t)rr.y * HH))[s];
    float4 r;
    r.x = ww.x * a.x + ww.y * b.x;
    r.y = ww.x * a.y + ww.y * b.y;
    r.z = ww.x * a.z + ww.y * b.z;
    r.w = ww.x * a.w + ww.y * b.w;
    out[idx] = r;
}

__global__ void moe_loss(const float* __restrict__ usage, float* __restrict__ out) {
    if (threadIdx.x == 0 && blockIdx.x == 0) {
        float u[EE], mx = -1e30f;
        for (int e = 0; e < EE; ++e) { u[e] = usage[e] / (float)TT; mx = fmaxf(mx, u[e]); }
        float se = 0.f;
        for (int e = 0; e < EE; ++e) se += __expf(u[e] - mx);
        float lse = logf(se) + mx;
        float uni = 1.f / (float)EE;
        float lu = logf(uni);
        float kl = 0.f;
        for (int e = 0; e < EE; ++e) kl += uni * (lu - (u[e] - lse));
        out[(size_t)TT * HH] = 0.01f * (kl / (float)EE);
    }
}

// ---------------------------------------------------------------------------
// launcher
// ---------------------------------------------------------------------------
extern "C" void kernel_launch(void* const* d_in, const int* in_sizes, int n_in,
                              void* d_out, int out_size, void* d_ws, size_t ws_size,
                              hipStream_t stream) {
    const float* x    = (const float*)d_in[0];   // [B,S,H]
    const float* gw   = (const float*)d_in[1];   // [H,E]
    const float* w1   = (const float*)d_in[2];   // [E,H,F]
    const float* b1   = (const float*)d_in[3];   // [E,F]
    const float* w2   = (const float*)d_in[4];   // [E,F,H]
    const float* b2   = (const float*)d_in[5];   // [E,H]
    float* out = (float*)d_out;

    char* base = (char*)d_ws;
    size_t o = 0;
    auto alloc = [&](size_t bytes) -> char* {
        char* p = base + o;
        o = (o + bytes + 255) & ~(size_t)255;
        return p;
    };
    __bf16* xb   = (__bf16*)alloc((size_t)TT * HH * 2);
    __bf16* w1t  = (__bf16*)alloc((size_t)EE * HH * FF * 2);
    __bf16* w2t  = (__bf16*)alloc((size_t)EE * FF * HH * 2);
    __bf16* hbuf = (__bf16*)alloc((size_t)NROWS * FF * 2);
    float*  eout = (float*)alloc((size_t)NROWS * HH * 4);
    int*    list = (int*)alloc((size_t)NROWS * 4);
    int2*   slot_row = (int2*)alloc((size_t)TT * sizeof(int2));
    int2*   tk_idx   = (int2*)alloc((size_t)TT * sizeof(int2));
    float2* tk_w     = (float2*)alloc((size_t)TT * sizeof(float2));
    int*    cnt = (int*)alloc(EE * 4);
    int*    off = (int*)alloc(EE * 4);
    int*    cur = (int*)alloc(EE * 4);
    float*  usage = (float*)alloc(EE * 4);
    (void)ws_size; (void)in_sizes; (void)n_in; (void)out_size;

    // 1) routing
    moe_init<<<1, 32, 0, stream>>>(cnt, usage);
    moe_gate<<<TT / 8, 256, 0, stream>>>(x, gw, cnt, usage, tk_idx, tk_w);
    moe_prefix<<<1, 32, 0, stream>>>(cnt, off, cur);
    moe_scatter<<<TT / 256, 256, 0, stream>>>(tk_idx, cur, list, slot_row);

    // 2) bf16 conversion (weights transposed for contiguous WMMA B fragments)
    cvt_x<<<(TT * HH / 4) / 256, 256, 0, stream>>>((const float4*)x, (v4bf*)xb);
    tcvt<<<dim3(FF / 32, HH / 32, EE), dim3(32, 8), 0, stream>>>(w1, w1t, HH, FF);
    tcvt<<<dim3(HH / 32, FF / 32, EE), dim3(32, 8), 0, stream>>>(w2, w2t, FF, HH);

    // 3) routed expert GEMMs (TDM-fed, double-buffered LDS, bf16 WMMA)
    moe_gemm1<<<dim3(FF / COLS, TT / 16, EE), 256, 0, stream>>>(xb, w1t, b1, list, cnt, off, hbuf);
    moe_gemm2<<<dim3(HH / COLS, TT / 16, EE), 256, 0, stream>>>(hbuf, w2t, b2, cnt, off, eout);

    // 4) weighted combine + aux loss
    moe_combine<<<(TT * HH / 4) / 256, 256, 0, stream>>>(eout, slot_row, tk_w, (float4*)out);
    moe_loss<<<1, 32, 0, stream>>>(usage, out);
}
